// Qwen3SelectiveModel_15685220565585
// MI455X (gfx1250) — matrix operand order repaired and checked by reading, not compiled, not run on hardware
//
#include <hip/hip_runtime.h>
#include <hip/hip_bf16.h>

typedef _Float16 v4h  __attribute__((ext_vector_type(4)));
typedef _Float16 v8h  __attribute__((ext_vector_type(8)));
typedef _Float16 v16h __attribute__((ext_vector_type(16)));
typedef float    v8f  __attribute__((ext_vector_type(8)));

namespace cfg {
constexpr int L   = 2;
constexpr int H   = 2048;
constexpr int NH  = 16;
constexpr int NKV = 8;
constexpr int HD  = 128;
constexpr int IFF = 6144;
constexpr int T   = 8192;
constexpr int B   = 8;
constexpr int S   = T / B;   // 1024 (reference fixes num_requests = 8)
}

// ---------------------------------------------------------------------------
// helpers
// ---------------------------------------------------------------------------
__device__ __forceinline__ v8f zero8() {
  v8f z = {0.f, 0.f, 0.f, 0.f, 0.f, 0.f, 0.f, 0.f};
  return z;
}

__device__ __forceinline__ v16h pack16(v8h lo, v8h hi) {
  union { v16h v; v8h h[2]; } u;
  u.h[0] = lo; u.h[1] = hi;
  return u.v;
}

// 16-bit A-matrix fragment (16x32 MxK): lane<16 holds K{0..7,16..23},
// lane>=16 holds K{8..15,24..31} of row M = lane%16 (ISA 7.12.2).
__device__ __forceinline__ v16h fragA(const _Float16* rp, int lane) {
  const int ks = (lane < 16) ? 0 : 8;
  return pack16(*(const v8h*)(rp + ks), *(const v8h*)(rp + ks + 16));
}

// 16-bit B-matrix fragment (32x16 KxN) from an N-major (transposed) tile:
// lane holds column N = lane%16, K = lane<16 ? {0..7,16..23} : {8..15,24..31}.
__device__ __forceinline__ v16h fragB(const _Float16* cp, int lane) {
  const int ks = (lane < 16) ? 0 : 8;
  return pack16(*(const v8h*)(cp + ks), *(const v8h*)(cp + ks + 16));
}

// CDNA5 async global -> LDS copy (16B per lane), tracked by ASYNCcnt.
// Generic LDS pointer is {aperture_hi, offset32}; low 32 bits are the
// wave-relative LDS byte offset the instruction's VDST operand expects.
__device__ __forceinline__ void async_copy_b128(const void* gptr, void* lptr) {
  const unsigned lds = (unsigned)(unsigned long long)lptr;
  const unsigned long long ga = (unsigned long long)gptr;
  asm volatile("global_load_async_to_lds_b128 %0, %1, off" ::"v"(lds), "v"(ga)
               : "memory");
}
template <int N>
__device__ __forceinline__ void wait_async() {
  asm volatile("s_wait_asynccnt %0" ::"i"(N) : "memory");
}

// ---------------------------------------------------------------------------
// Weight pre-pass: W_f32[K,N] -> Wt_f16[N,K]  (32x32 tiles through LDS,
// coalesced b128 loads, coalesced b64 stores)
// ---------------------------------------------------------------------------
__global__ __launch_bounds__(256) void k_w_to_f16t(const float* __restrict__ W,
                                                   _Float16* __restrict__ Wt,
                                                   int K, int N) {
  __shared__ __attribute__((aligned(16))) _Float16 tile[32][32 + 8];
  const int tid = threadIdx.x;
  const int n0 = blockIdx.x * 32, k0 = blockIdx.y * 32;
  const int r = tid >> 3, c = (tid & 7) * 4;
  {
    const float4 v = *(const float4*)(W + (long)(k0 + r) * N + n0 + c);
    tile[c + 0][r] = (_Float16)v.x;
    tile[c + 1][r] = (_Float16)v.y;
    tile[c + 2][r] = (_Float16)v.z;
    tile[c + 3][r] = (_Float16)v.w;
  }
  __syncthreads();
  *(v4h*)(Wt + (long)(n0 + r) * K + k0 + c) = *(const v4h*)&tile[r][c];
}

// ---------------------------------------------------------------------------
// RMSNorm (row-wise over Hn), f16 or f32 output
// ---------------------------------------------------------------------------
template <bool OUT_HALF>
__global__ __launch_bounds__(256) void k_rmsnorm(const float* __restrict__ x,
                                                 const float* __restrict__ w,
                                                 void* __restrict__ outp, int Hn) {
  const int row = blockIdx.x;
  const float* xr = x + (long)row * Hn;
  float ss = 0.f;
  for (int i = threadIdx.x; i < Hn; i += 256) { float v = xr[i]; ss += v * v; }
  for (int off = 16; off > 0; off >>= 1) ss += __shfl_xor(ss, off, 32);
  __shared__ float red[8];
  if ((threadIdx.x & 31) == 0) red[threadIdx.x >> 5] = ss;
  __syncthreads();
  if (threadIdx.x == 0) {
    float t = 0.f;
    for (int i = 0; i < 8; ++i) t += red[i];
    red[0] = rsqrtf(t / (float)Hn + 1e-6f);
  }
  __syncthreads();
  const float rstd = red[0];
  if (OUT_HALF) {
    _Float16* o = (_Float16*)outp + (long)row * Hn;
    for (int i = threadIdx.x; i < Hn; i += 256) o[i] = (_Float16)(xr[i] * rstd * w[i]);
  } else {
    float* o = (float*)outp + (long)row * Hn;
    for (int i = threadIdx.x; i < Hn; i += 256) o[i] = xr[i] * rstd * w[i];
  }
}

// ---------------------------------------------------------------------------
// WMMA GEMM: C[M,N] = A_f16[M,K] * Wt_f16[N,K]^T
// MODE: 0 = store f32, 1 = store f16, 2 = out = resid + C (f32, in-place x)
// Block tile 128x128, BK=32, 8 waves (4x2), wave tile 32x64 (2x4 WMMA ->
// 8 WMMAs per 6 fragment loads).  Async global->LDS double buffering:
// tile t+1 streams in via GLOBAL_LOAD_ASYNC_TO_LDS_B128 while WMMAs consume
// tile t; s_wait_asynccnt 4 waits only for tile t's copies.
// ---------------------------------------------------------------------------
template <int MODE>
__global__ __launch_bounds__(256) void k_gemm_f16(const _Float16* __restrict__ A,
                                                  const _Float16* __restrict__ Wt,
                                                  void* __restrict__ outp,
                                                  const float* __restrict__ resid,
                                                  int M, int N, int K) {
  constexpr int BM = 128, BN = 128, BK = 32, PAD = 8;
  __shared__ __attribute__((aligned(16))) _Float16 As[2][BM][BK + PAD];
  __shared__ __attribute__((aligned(16))) _Float16 Bs[2][BN][BK + PAD];
  const int tid = threadIdx.x, lane = tid & 31, wave = tid >> 5;
  const int wm = wave & 3, wn = wave >> 2;  // wave tile: rows wm*32, cols wn*64
  const int m0 = blockIdx.y * BM, n0 = blockIdx.x * BN;

  v8f acc[2][4];
#pragma unroll
  for (int i = 0; i < 2; ++i)
#pragma unroll
    for (int j = 0; j < 4; ++j) acc[i][j] = zero8();

  const int ar = tid >> 1, ac = (tid & 1) * 16;  // A: 128 rows x 32, 16 halves/thr
  const _Float16* gA = A + (long)(m0 + ar) * K + ac;
  const _Float16* gB = Wt + (long)(n0 + ar) * K + ac;  // B: same 128x32 shape

  auto issue_tile = [&](int buf, int k0) {
    async_copy_b128(gA + k0, &As[buf][ar][ac]);
    async_copy_b128(gA + k0 + 8, &As[buf][ar][ac + 8]);
    async_copy_b128(gB + k0, &Bs[buf][ar][ac]);
    async_copy_b128(gB + k0 + 8, &Bs[buf][ar][ac + 8]);
  };

  const int nk = K / BK;
  issue_tile(0, 0);

  for (int kt = 0; kt < nk; ++kt) {
    const int buf = kt & 1;
    if (kt + 1 < nk) {
      issue_tile(buf ^ 1, (kt + 1) * BK);
      if (kt + 2 < nk) {
        __builtin_prefetch(gA + (kt + 2) * BK, 0, 1);
        __builtin_prefetch(gB + (kt + 2) * BK, 0, 1);
      }
      wait_async<4>();  // tile `buf` complete; tile buf^1 still in flight
    } else {
      wait_async<0>();
    }
    __syncthreads();
    const v16h a0 = fragA(&As[buf][wm * 32 + (lane & 15)][0], lane);
    const v16h a1 = fragA(&As[buf][wm * 32 + 16 + (lane & 15)][0], lane);
#pragma unroll
    for (int sn = 0; sn < 4; ++sn) {
      const v16h b = fragB(&Bs[buf][wn * 64 + sn * 16 + (lane & 15)][0], lane);
      acc[0][sn] = __builtin_amdgcn_wmma_f32_16x16x32_f16(false, a0, false, b, (short)0, acc[0][sn], false, false);
      acc[1][sn] = __builtin_amdgcn_wmma_f32_16x16x32_f16(false, a1, false, b, (short)0, acc[1][sn], false, false);
    }
    __syncthreads();  // all waves done reading `buf` before it is re-filled
  }

#pragma unroll
  for (int sm = 0; sm < 2; ++sm)
#pragma unroll
    for (int sn = 0; sn < 4; ++sn) {
      const int col = n0 + wn * 64 + sn * 16 + (lane & 15);
      const int rbase = m0 + wm * 32 + sm * 16 + ((lane >> 4) << 3);
#pragma unroll
      for (int r = 0; r < 8; ++r) {
        const long idx = (long)(rbase + r) * N + col;
        const float v = acc[sm][sn][r];
        if (MODE == 0)      ((float*)outp)[idx] = v;
        else if (MODE == 1) ((_Float16*)outp)[idx] = (_Float16)v;
        else                ((float*)outp)[idx] = resid[idx] + v;
      }
    }
}

// ---------------------------------------------------------------------------
// Fused gate/up GEMM:  out_f16 = silu(A@Wg) * (A@Wu), both weights f16 [N,K]
// Block 128x64, wave 32x32 per weight (8 WMMAs per K-step).
// ---------------------------------------------------------------------------
__global__ __launch_bounds__(256) void k_gemm_gateup(const _Float16* __restrict__ A,
                                                     const _Float16* __restrict__ Wgt,
                                                     const _Float16* __restrict__ Wut,
                                                     _Float16* __restrict__ outp,
                                                     int M, int N, int K) {
  constexpr int BM = 128, BN = 64, BK = 32, PAD = 8;
  __shared__ __attribute__((aligned(16))) _Float16 As[2][BM][BK + PAD];
  __shared__ __attribute__((aligned(16))) _Float16 Bg[2][BN][BK + PAD];
  __shared__ __attribute__((aligned(16))) _Float16 Bu[2][BN][BK + PAD];
  const int tid = threadIdx.x, lane = tid & 31, wave = tid >> 5;
  const int wm = wave & 3, wn = wave >> 2;
  const int m0 = blockIdx.y * BM, n0 = blockIdx.x * BN;

  v8f ag[2][2], au[2][2];
#pragma unroll
  for (int i = 0; i < 2; ++i)
#pragma unroll
    for (int j = 0; j < 2; ++j) { ag[i][j] = zero8(); au[i][j] = zero8(); }

  const int ar = tid >> 1, ac = (tid & 1) * 16;
  const int br = tid >> 2, bc = (tid & 3) * 8;
  const _Float16* gA = A + (long)(m0 + ar) * K + ac;
  const _Float16* gG = Wgt + (long)(n0 + br) * K + bc;
  const _Float16* gU = Wut + (long)(n0 + br) * K + bc;

  auto issue_tile = [&](int buf, int k0) {
    async_copy_b128(gA + k0, &As[buf][ar][ac]);
    async_copy_b128(gA + k0 + 8, &As[buf][ar][ac + 8]);
    async_copy_b128(gG + k0, &Bg[buf][br][bc]);
    async_copy_b128(gU + k0, &Bu[buf][br][bc]);
  };

  const int nk = K / BK;
  issue_tile(0, 0);

  for (int kt = 0; kt < nk; ++kt) {
    const int buf = kt & 1;
    if (kt + 1 < nk) {
      issue_tile(buf ^ 1, (kt + 1) * BK);
      wait_async<4>();
    } else {
      wait_async<0>();
    }
    __syncthreads();
    const v16h a0 = fragA(&As[buf][wm * 32 + (lane & 15)][0], lane);
    const v16h a1 = fragA(&As[buf][wm * 32 + 16 + (lane & 15)][0], lane);
    const v16h b0 = fragB(&Bg[buf][wn * 32 + (lane & 15)][0], lane);
    const v16h b1 = fragB(&Bg[buf][wn * 32 + 16 + (lane & 15)][0], lane);
    const v16h c0 = fragB(&Bu[buf][wn * 32 + (lane & 15)][0], lane);
    const v16h c1 = fragB(&Bu[buf][wn * 32 + 16 + (lane & 15)][0], lane);
    ag[0][0] = __builtin_amdgcn_wmma_f32_16x16x32_f16(false, a0, false, b0, (short)0, ag[0][0], false, false);
    ag[0][1] = __builtin_amdgcn_wmma_f32_16x16x32_f16(false, a0, false, b1, (short)0, ag[0][1], false, false);
    ag[1][0] = __builtin_amdgcn_wmma_f32_16x16x32_f16(false, a1, false, b0, (short)0, ag[1][0], false, false);
    ag[1][1] = __builtin_amdgcn_wmma_f32_16x16x32_f16(false, a1, false, b1, (short)0, ag[1][1], false, false);
    au[0][0] = __builtin_amdgcn_wmma_f32_16x16x32_f16(false, a0, false, c0, (short)0, au[0][0], false, false);
    au[0][1] = __builtin_amdgcn_wmma_f32_16x16x32_f16(false, a0, false, c1, (short)0, au[0][1], false, false);
    au[1][0] = __builtin_amdgcn_wmma_f32_16x16x32_f16(false, a1, false, c0, (short)0, au[1][0], false, false);
    au[1][1] = __builtin_amdgcn_wmma_f32_16x16x32_f16(false, a1, false, c1, (short)0, au[1][1], false, false);
    __syncthreads();
  }

#pragma unroll
  for (int sm = 0; sm < 2; ++sm)
#pragma unroll
    for (int sn = 0; sn < 2; ++sn) {
      const int col = n0 + wn * 32 + sn * 16 + (lane & 15);
      const int rbase = m0 + wm * 32 + sm * 16 + ((lane >> 4) << 3);
#pragma unroll
      for (int r = 0; r < 8; ++r) {
        const float g = ag[sm][sn][r];
        const float u = au[sm][sn][r];
        const float s = g / (1.f + __expf(-g));
        outp[(long)(rbase + r) * N + col] = (_Float16)(s * u);
      }
    }
}

// ---------------------------------------------------------------------------
// Per-head q/k RMSNorm + RoPE.  One wave per (token, head); heads 0..15 = q,
// 16..23 = k.  HD=128: lane holds dims {lane, lane+32, lane+64, lane+96};
// RoPE pairs (d, d+64) stay within a lane.
// ---------------------------------------------------------------------------
__global__ __launch_bounds__(256) void k_qknorm_rope(const float* __restrict__ q32,
                                                     const float* __restrict__ k32,
                                                     const float* __restrict__ qw,
                                                     const float* __restrict__ kw,
                                                     _Float16* __restrict__ qh,
                                                     _Float16* __restrict__ kh,
                                                     const int* __restrict__ nreq) {
  const int lane = threadIdx.x & 31;
  const int gw = (blockIdx.x * blockDim.x + threadIdx.x) >> 5;
  const int t = gw / (cfg::NH + cfg::NKV);
  const int h = gw % (cfg::NH + cfg::NKV);
  const bool isq = h < cfg::NH;
  const float* src = isq ? q32 + (long)t * (cfg::NH * cfg::HD) + h * cfg::HD
                         : k32 + (long)t * (cfg::NKV * cfg::HD) + (h - cfg::NH) * cfg::HD;
  _Float16* dst = isq ? qh + (long)t * (cfg::NH * cfg::HD) + h * cfg::HD
                      : kh + (long)t * (cfg::NKV * cfg::HD) + (h - cfg::NH) * cfg::HD;
  const float* w = isq ? qw : kw;

  float x0 = src[lane], x1 = src[lane + 32], x2 = src[lane + 64], x3 = src[lane + 96];
  float ss = x0 * x0 + x1 * x1 + x2 * x2 + x3 * x3;
  for (int off = 16; off > 0; off >>= 1) ss += __shfl_xor(ss, off, 32);
  const float rstd = rsqrtf(ss / 128.f + 1e-6f);
  x0 = x0 * rstd * w[lane];
  x1 = x1 * rstd * w[lane + 32];
  x2 = x2 * rstd * w[lane + 64];
  x3 = x3 * rstd * w[lane + 96];

  const int Sloc = cfg::T / nreq[0];
  const float pos = (float)(t % Sloc);
  const float kln = -13.815510557964274f / 64.f;  // -ln(1e6)/(HD/2)
  const float f0 = pos * __expf(kln * (float)lane);
  const float f1 = pos * __expf(kln * (float)(lane + 32));
  float c0, s0, c1, s1;
  __sincosf(f0, &s0, &c0);
  __sincosf(f1, &s1, &c1);
  dst[lane]      = (_Float16)(x0 * c0 - x2 * s0);
  dst[lane + 32] = (_Float16)(x1 * c1 - x3 * s1);
  dst[lane + 64] = (_Float16)(x2 * c0 + x0 * s0);
  dst[lane + 96] = (_Float16)(x3 * c1 + x1 * s1);
}

// ---------------------------------------------------------------------------
// Flash-style causal GQA attention.  Grid: (S/64, NH, B), 256 threads.
// 64-row Q block; 32-key blocks; WMMA for both Q*K^T and P*V.
// Q and K tiles stream in via async global->LDS.
// ---------------------------------------------------------------------------
__global__ __launch_bounds__(256) void k_attention(const _Float16* __restrict__ qh,
                                                   const _Float16* __restrict__ kh,
                                                   const _Float16* __restrict__ vh,
                                                   _Float16* __restrict__ outp) {
  constexpr int BQ = 64, BKV = 32, HD = cfg::HD, PAD = 8;
  __shared__ __attribute__((aligned(16))) _Float16 Qs[BQ][HD + PAD];
  __shared__ __attribute__((aligned(16))) _Float16 Ks[BKV][HD + PAD];
  __shared__ __attribute__((aligned(16))) _Float16 Vt[HD][BKV + PAD];  // HD-major V^T
  __shared__ float Ss[BQ][BKV + 2];
  __shared__ __attribute__((aligned(16))) _Float16 Ps[BQ][BKV + PAD];
  __shared__ float rmaxs[BQ], rsums[BQ], rscale[BQ];

  const int tid = threadIdx.x, lane = tid & 31, wave = tid >> 5;
  const int q0 = blockIdx.x * BQ;
  const int head = blockIdx.y, batch = blockIdx.z;
  const int kvh = head >> 1;  // NH/NKV == 2

  // Q block (64 x 128 halves): async global -> LDS
  {
    const int r = tid >> 2, c = (tid & 3) * 32;
    const _Float16* g = qh + (long)(batch * cfg::S + q0 + r) * (cfg::NH * HD) + head * HD + c;
    async_copy_b128(g, &Qs[r][c]);
    async_copy_b128(g + 8, &Qs[r][c + 8]);
    async_copy_b128(g + 16, &Qs[r][c + 16]);
    async_copy_b128(g + 24, &Qs[r][c + 24]);
  }
  if (tid < BQ) { rmaxs[tid] = -3.0e38f; rsums[tid] = 0.f; }

  v8f o[4];
#pragma unroll
  for (int j = 0; j < 4; ++j) o[j] = zero8();

  const int om = wave & 3, on = (wave >> 2) * 64;  // output tiling (4 x 2 waves)
  const int sm = wave & 3, sn = wave >> 2;         // score tiling (4 x 2 tiles)
  const float scale = 0.08838834764831845f;        // 1/sqrt(128)
  const int nkb = (q0 >> 5) + 2;                   // causal: key blocks needed

  for (int kb = 0; kb < nkb; ++kb) {
    const int kv0 = kb * BKV;
    // K block (32 x 128): async global -> LDS
    {
      const int r = tid >> 3, c = (tid & 7) * 16;
      const _Float16* g = kh + (long)(batch * cfg::S + kv0 + r) * (cfg::NKV * HD) + kvh * HD + c;
      async_copy_b128(g, &Ks[r][c]);
      async_copy_b128(g + 8, &Ks[r][c + 8]);
    }
    // V block transposed (HD-major): two b128 loads, then b16 scatter
    {
      const int i = tid & 31, c8 = (tid >> 5) * 16;
      const _Float16* g = vh + (long)(batch * cfg::S + kv0 + i) * (cfg::NKV * HD) + kvh * HD + c8;
      v8h t0 = *(const v8h*)g, t1 = *(const v8h*)(g + 8);
#pragma unroll
      for (int j = 0; j < 8; ++j) Vt[c8 + j][i] = t0[j];
#pragma unroll
      for (int j = 0; j < 8; ++j) Vt[c8 + 8 + j][i] = t1[j];
    }
    wait_async<0>();
    __syncthreads();

    // scores: each wave one 16x16 tile of the 64x32 block
    v8f sacc = zero8();
    {
      const _Float16* arow = &Qs[sm * 16 + (lane & 15)][0];
      const _Float16* brow = &Ks[sn * 16 + (lane & 15)][0];
#pragma unroll
      for (int kq = 0; kq < HD; kq += 32) {
        const v16h a = fragA(arow + kq, lane);
        const v16h b = fragB(brow + kq, lane);
        sacc = __builtin_amdgcn_wmma_f32_16x16x32_f16(false, a, false, b, (short)0, sacc, false, false);
      }
    }
#pragma unroll
    for (int r = 0; r < 8; ++r) {
      const int row = sm * 16 + ((lane >> 4) << 3) + r;
      const int col = sn * 16 + (lane & 15);
      float v = sacc[r] * scale;
      if (kv0 + col > q0 + row) v = -3.0e38f;  // causal mask
      Ss[row][col] = v;
    }
    __syncthreads();

    // online softmax per row (threads 0..63)
    if (tid < BQ) {
      const float m = rmaxs[tid];
      float mb = -3.0e38f;
#pragma unroll
      for (int c = 0; c < BKV; ++c) mb = fmaxf(mb, Ss[tid][c]);
      const float mn = fmaxf(m, mb);
      const float sc = __expf(m - mn);
      float s = 0.f;
#pragma unroll
      for (int c = 0; c < BKV; ++c) {
        const float p = __expf(Ss[tid][c] - mn);
        Ps[tid][c] = (_Float16)p;
        s += p;
      }
      rscale[tid] = sc;
      rsums[tid] = rsums[tid] * sc + s;
      rmaxs[tid] = mn;
    }
    __syncthreads();

    // rescale O accumulators, then O += P * V
    {
      const int rb = om * 16 + ((lane >> 4) << 3);
#pragma unroll
      for (int r = 0; r < 8; ++r) {
        const float f = rscale[rb + r];
#pragma unroll
        for (int j = 0; j < 4; ++j) o[j][r] *= f;
      }
      const v16h ap = fragA(&Ps[om * 16 + (lane & 15)][0], lane);
#pragma unroll
      for (int j = 0; j < 4; ++j) {
        const v16h bv = fragB(&Vt[on + j * 16 + (lane & 15)][0], lane);
        o[j] = __builtin_amdgcn_wmma_f32_16x16x32_f16(false, ap, false, bv, (short)0, o[j], false, false);
      }
    }
    __syncthreads();
  }

  // epilogue: divide by row sums, store f16 [t][head*HD + hd]
  {
    const int rb = om * 16 + ((lane >> 4) << 3);
#pragma unroll
    for (int r = 0; r < 8; ++r) {
      const float inv = 1.f / rsums[rb + r];
      const long trow = (long)(batch * cfg::S + q0 + rb + r);
#pragma unroll
      for (int j = 0; j < 4; ++j) {
        const int col = head * HD + on + j * 16 + (lane & 15);
        outp[trow * (cfg::NH * HD) + col] = (_Float16)(o[j][r] * inv);
      }
    }
  }
}

// ---------------------------------------------------------------------------
// host orchestration
// ---------------------------------------------------------------------------
extern "C" void kernel_launch(void* const* d_in, const int* in_sizes, int n_in,
                              void* d_out, int out_size, void* d_ws, size_t ws_size,
                              hipStream_t stream) {
  using namespace cfg;
  const float* hidden = (const float*)d_in[0];
  const float* Wq  = (const float*)d_in[1];
  const float* Wk  = (const float*)d_in[2];
  const float* Wv  = (const float*)d_in[3];
  const float* Wo  = (const float*)d_in[4];
  const float* qnw = (const float*)d_in[5];
  const float* knw = (const float*)d_in[6];
  const float* ln1 = (const float*)d_in[7];
  const float* ln2 = (const float*)d_in[8];
  const float* Wg  = (const float*)d_in[9];
  const float* Wu  = (const float*)d_in[10];
  const float* Wd  = (const float*)d_in[11];
  const float* nw  = (const float*)d_in[12];
  const int*   nreq = (const int*)d_in[13];
  (void)in_sizes; (void)n_in; (void)out_size; (void)ws_size;

  char* ws = (char*)d_ws;
  size_t off = 0;
  auto carve = [&](size_t bytes) -> void* {
    void* p = ws + off;
    off = (off + bytes + 255) & ~(size_t)255;
    return p;
  };
  float*     x    = (float*)carve((size_t)T * H * 4);            //  67 MB
  _Float16*  hn   = (_Float16*)carve((size_t)T * H * 2);         //  33 MB
  float*     q32  = (float*)carve((size_t)T * NH * HD * 4);      //  67 MB
  float*     k32  = (float*)carve((size_t)T * NKV * HD * 4);     //  33 MB
  _Float16*  qf   = (_Float16*)carve((size_t)T * NH * HD * 2);   //  33 MB
  _Float16*  kf   = (_Float16*)carve((size_t)T * NKV * HD * 2);  //  17 MB
  _Float16*  vf   = (_Float16*)carve((size_t)T * NKV * HD * 2);  //  17 MB
  _Float16*  attn = (_Float16*)carve((size_t)T * NH * HD * 2);   //  33 MB
  _Float16*  act  = (_Float16*)carve((size_t)T * IFF * 2);       // 101 MB
  _Float16*  wt_a = (_Float16*)carve((size_t)IFF * H * 2);       //  25 MB (largest W)
  _Float16*  wt_b = (_Float16*)carve((size_t)IFF * H * 2);       //  25 MB

  hipMemcpyAsync(x, hidden, (size_t)T * H * 4, hipMemcpyDeviceToDevice, stream);

  const dim3 blk(256);
  const dim3 g_h(H / 128, T / 128);            // GEMM to N=2048
  const dim3 g_kv((NKV * HD) / 128, T / 128);  // GEMM to N=1024
  const dim3 g_mlp(IFF / 64, T / 128);         // gate/up, N=6144
  const dim3 g_attn(S / 64, NH, B);

  auto transpose = [&](const float* W, _Float16* Wt, int K, int N) {
    k_w_to_f16t<<<dim3(N / 32, K / 32), blk, 0, stream>>>(W, Wt, K, N);
  };

  for (int l = 0; l < L; ++l) {
    const float* Wq_l = Wq + (size_t)l * H * (NH * HD);
    const float* Wk_l = Wk + (size_t)l * H * (NKV * HD);
    const float* Wv_l = Wv + (size_t)l * H * (NKV * HD);
    const float* Wo_l = Wo + (size_t)l * (NH * HD) * H;
    const float* Wg_l = Wg + (size_t)l * H * IFF;
    const float* Wu_l = Wu + (size_t)l * H * IFF;
    const float* Wd_l = Wd + (size_t)l * IFF * H;

    // x -> rmsnorm(ln1) -> hn (f16)
    k_rmsnorm<true><<<T, blk, 0, stream>>>(x, ln1 + (size_t)l * H, hn, H);
    // q/k (f32 for per-head norm + rope), v directly f16
    transpose(Wq_l, wt_a, H, NH * HD);
    k_gemm_f16<0><<<g_h, blk, 0, stream>>>(hn, wt_a, q32, nullptr, T, NH * HD, H);
    transpose(Wk_l, wt_a, H, NKV * HD);
    k_gemm_f16<0><<<g_kv, blk, 0, stream>>>(hn, wt_a, k32, nullptr, T, NKV * HD, H);
    transpose(Wv_l, wt_a, H, NKV * HD);
    k_gemm_f16<1><<<g_kv, blk, 0, stream>>>(hn, wt_a, vf, nullptr, T, NKV * HD, H);
    // per-head rmsnorm + rope -> f16 q/k
    k_qknorm_rope<<<(T * (NH + NKV)) / 8, blk, 0, stream>>>(
        q32, k32, qnw + (size_t)l * HD, knw + (size_t)l * HD, qf, kf, nreq);
    // attention
    k_attention<<<g_attn, blk, 0, stream>>>(qf, kf, vf, attn);
    // x += attn @ Wo
    transpose(Wo_l, wt_a, NH * HD, H);
    k_gemm_f16<2><<<g_h, blk, 0, stream>>>(attn, wt_a, x, x, T, H, NH * HD);
    // x -> rmsnorm(ln2) -> hn
    k_rmsnorm<true><<<T, blk, 0, stream>>>(x, ln2 + (size_t)l * H, hn, H);
    // act = silu(hn@Wg) * (hn@Wu)
    transpose(Wg_l, wt_a, H, IFF);
    transpose(Wu_l, wt_b, H, IFF);
    k_gemm_gateup<<<g_mlp, blk, 0, stream>>>(hn, wt_a, wt_b, act, T, IFF, H);
    // x += act @ Wd
    transpose(Wd_l, wt_a, IFF, H);
    k_gemm_f16<2><<<g_h, blk, 0, stream>>>(act, wt_a, x, x, T, H, IFF);
  }

  // final norm -> f32 output
  k_rmsnorm<false><<<T, blk, 0, stream>>>(x, nw, d_out, H);
}
